// FSCILGate_19688130085038
// MI455X (gfx1250) — compile-verified, matrix operand order checked by reading, not run on hardware
//
#include <hip/hip_runtime.h>

// MoE gate: logits = x @ W^T + b, softmax(E=64), top-2, column-normalized * capacity, aux loss.
// GEMM uses V_WMMA_F32_16X16X32_BF16 with bf16 hi/lo split (3 WMMAs ~ fp32 accuracy).
// x traffic (512 MB @ 23.3 TB/s ~ 22us) is the floor. The K-loop is double-buffered
// (loads for k+32 issued before WMMAs of k), x is streamed with non-temporal loads
// (read-once data; protects W in WGP$/L2), and the A stream is prefetched ~1KB ahead.

#define DIMK  4096
#define NEXP  64
#define NROWS 32768
#define CAPF  40960.0f   // int(1.25 * 32768)

typedef __attribute__((ext_vector_type(16))) __bf16 v16bf;
typedef __attribute__((ext_vector_type(8)))  float  v8f;
typedef __attribute__((ext_vector_type(4)))  float  v4f;   // clang vector: OK for nontemporal builtin

union FragB { v16bf v; uint4 q[2]; };

// ---------------- prep: split W (f32) into bf16 hi/lo; zero global accumulators ----------------
__global__ __launch_bounds__(256) void moe_prep_kernel(const float* __restrict__ W,
                                                       unsigned short* __restrict__ whi_u,
                                                       unsigned short* __restrict__ wlo_u,
                                                       float* __restrict__ accums /*192 f32*/) {
  int idx = blockIdx.x * 256 + threadIdx.x;
  __bf16* whi = (__bf16*)whi_u;
  __bf16* wlo = (__bf16*)wlo_u;
  if (idx < NEXP * DIMK) {
    float v = W[idx];
    __bf16 h = (__bf16)v;
    whi[idx] = h;
    wlo[idx] = (__bf16)(v - (float)h);
  }
  if (blockIdx.x == 0 && threadIdx.x < 192) accums[threadIdx.x] = 0.0f;
}

// ---------------- main: WMMA GEMM + softmax + top2 + partial reductions ----------------
__global__ __launch_bounds__(256) void moe_gate_kernel(
    const float* __restrict__ x, const float* __restrict__ bias,
    const unsigned short* __restrict__ whi_u, const unsigned short* __restrict__ wlo_u,
    float* __restrict__ gDen, float* __restrict__ gImp, float* __restrict__ gCnt,
    int* __restrict__ oI0, int* __restrict__ oI1,
    float* __restrict__ oS0, float* __restrict__ oS1)
{
  __shared__ float sLog[128 * 68];   // 128 rows x 64 experts, stride 68 (bank-conflict pad)
  __shared__ float sDen[64];
  __shared__ float sImp[64];
  __shared__ float sCnt[64];

  const int tid = threadIdx.x;
  if (tid < 64) { sDen[tid] = 0.f; sImp[tid] = 0.f; sCnt[tid] = 0.f; }

  const int wave = tid >> 5;
  const int lane = tid & 31;
  const int lmod = lane & 15;       // row within A tile / col within B,C tiles
  const int lhi  = lane >> 4;       // K-half selector (A/B), row+8 selector (C)
  const int rowBase = blockIdx.x * 128 + wave * 16;

  const float*  arow = x + (size_t)(rowBase + lmod) * DIMK + lhi * 8;
  const __bf16* Whi  = (const __bf16*)whi_u;
  const __bf16* Wlo  = (const __bf16*)wlo_u;
  const size_t  wrow = (size_t)lmod * DIMK + lhi * 8;

  v8f acc0 = {}, acc1 = {}, acc2 = {}, acc3 = {};

  // A stream: each 128B chunk of x is read exactly once -> non-temporal (protect W in caches)
  #define LOAD_A(K, A0, A1, A2, A3) do {                                        \
    A0 = __builtin_nontemporal_load((const v4f*)(arow + (K)));                  \
    A1 = __builtin_nontemporal_load((const v4f*)(arow + (K) + 4));              \
    A2 = __builtin_nontemporal_load((const v4f*)(arow + (K) + 16));             \
    A3 = __builtin_nontemporal_load((const v4f*)(arow + (K) + 20));             \
  } while (0)

  #define LOAD_B(K, NT, BH, BL) do {                                            \
    const __bf16* wh_ = Whi + wrow + (size_t)(NT) * 16 * DIMK + (K);            \
    const __bf16* wl_ = Wlo + wrow + (size_t)(NT) * 16 * DIMK + (K);            \
    BH.q[0] = *(const uint4*)(wh_);      BH.q[1] = *(const uint4*)(wh_ + 16);   \
    BL.q[0] = *(const uint4*)(wl_);      BL.q[1] = *(const uint4*)(wl_ + 16);   \
  } while (0)

  #define CVT_A(A0, A1, A2, A3, AHI, ALO) do {                                  \
    float av_[16] = {A0.x,A0.y,A0.z,A0.w, A1.x,A1.y,A1.z,A1.w,                  \
                     A2.x,A2.y,A2.z,A2.w, A3.x,A3.y,A3.z,A3.w};                 \
    _Pragma("unroll")                                                           \
    for (int i_ = 0; i_ < 16; ++i_) {                                           \
      __bf16 h_ = (__bf16)av_[i_];                                              \
      AHI.v[i_] = h_;                                                           \
      ALO.v[i_] = (__bf16)(av_[i_] - (float)h_);                                \
    }                                                                           \
  } while (0)

  #define WMMA3(ACC, AH, AL, BH, BL) do {                                       \
    ACC = __builtin_amdgcn_wmma_f32_16x16x32_bf16(false, AH.v, false, BH.v,     \
                                                  (short)0, ACC, false, false); \
    ACC = __builtin_amdgcn_wmma_f32_16x16x32_bf16(false, AL.v, false, BH.v,     \
                                                  (short)0, ACC, false, false); \
    ACC = __builtin_amdgcn_wmma_f32_16x16x32_bf16(false, AH.v, false, BL.v,     \
                                                  (short)0, ACC, false, false); \
  } while (0)

  // ---- software-pipelined K loop: loads for k+32 issued before WMMAs of k ----
  v4f a0, a1, a2, a3;
  FragB bh0, bl0, bh1, bl1, bh2, bl2, bh3, bl3;
  LOAD_A(0, a0, a1, a2, a3);
  LOAD_B(0, 0, bh0, bl0);
  LOAD_B(0, 1, bh1, bl1);
  LOAD_B(0, 2, bh2, bl2);
  LOAD_B(0, 3, bh3, bl3);

  for (int k = 0; k < DIMK - 32; k += 32) {
    v4f na0, na1, na2, na3;
    FragB nbh0, nbl0, nbh1, nbl1, nbh2, nbl2, nbh3, nbl3;
    LOAD_A(k + 32, na0, na1, na2, na3);
    LOAD_B(k + 32, 0, nbh0, nbl0);
    LOAD_B(k + 32, 1, nbh1, nbl1);
    LOAD_B(k + 32, 2, nbh2, nbl2);
    LOAD_B(k + 32, 3, nbh3, nbl3);

    // deep prefetch of the A stream (~8 tiles ahead), branchless clamp
    int kp = k + 256;
    if (kp > DIMK - 32) kp = DIMK - 32;
    __builtin_prefetch(arow + kp, 0, 3);

    FragB ahi, alo;
    CVT_A(a0, a1, a2, a3, ahi, alo);
    WMMA3(acc0, ahi, alo, bh0, bl0);
    WMMA3(acc1, ahi, alo, bh1, bl1);
    WMMA3(acc2, ahi, alo, bh2, bl2);
    WMMA3(acc3, ahi, alo, bh3, bl3);

    a0 = na0; a1 = na1; a2 = na2; a3 = na3;
    bh0 = nbh0; bl0 = nbl0; bh1 = nbh1; bl1 = nbl1;
    bh2 = nbh2; bl2 = nbl2; bh3 = nbh3; bl3 = nbl3;
  }
  {
    FragB ahi, alo;
    CVT_A(a0, a1, a2, a3, ahi, alo);
    WMMA3(acc0, ahi, alo, bh0, bl0);
    WMMA3(acc1, ahi, alo, bh1, bl1);
    WMMA3(acc2, ahi, alo, bh2, bl2);
    WMMA3(acc3, ahi, alo, bh3, bl3);
  }
  #undef LOAD_A
  #undef LOAD_B
  #undef CVT_A
  #undef WMMA3

  // Epilogue: add bias, spill logits to LDS (C layout: VGPR r -> row r + 8*lhi, col lmod)
  {
    float b0 = bias[ 0 + lmod];
    float b1 = bias[16 + lmod];
    float b2 = bias[32 + lmod];
    float b3 = bias[48 + lmod];
    const int rbase = wave * 16 + lhi * 8;
    #pragma unroll
    for (int r = 0; r < 8; ++r) {
      int rowL = rbase + r;
      sLog[rowL * 68 +  0 + lmod] = acc0[r] + b0;
      sLog[rowL * 68 + 16 + lmod] = acc1[r] + b1;
      sLog[rowL * 68 + 32 + lmod] = acc2[r] + b2;
      sLog[rowL * 68 + 48 + lmod] = acc3[r] + b3;
    }
  }
  __syncthreads();

  // Per-row phase: softmax + top-2 + per-expert LDS accumulation (threads 0..127 = rows)
  if (tid < 128) {
    float* lrow = &sLog[tid * 68];
    float r[64];
    #pragma unroll
    for (int j = 0; j < 16; ++j) {
      v4f q = *(const v4f*)(lrow + j * 4);
      r[j*4+0] = q.x; r[j*4+1] = q.y; r[j*4+2] = q.z; r[j*4+3] = q.w;
    }
    float m = -3.4e38f;
    #pragma unroll
    for (int e = 0; e < 64; ++e) m = fmaxf(m, r[e]);
    float sum = 0.f;
    #pragma unroll
    for (int e = 0; e < 64; ++e) { r[e] = __expf(r[e] - m); sum += r[e]; }
    float inv = 1.0f / sum;

    float v0 = -1.f, v1 = -1.f; int i0 = 0, i1 = 0;   // exp() > 0 always
    #pragma unroll
    for (int e = 0; e < 64; ++e) {
      float val = r[e];
      if (val > v0)      { v1 = v0; i1 = i0; v0 = val; i0 = e; }
      else if (val > v1) { v1 = val; i1 = e; }
    }
    float s0 = v0 * inv, s1 = v1 * inv;

    int grow = blockIdx.x * 128 + tid;
    oI0[grow] = i0; oI1[grow] = i1; oS0[grow] = s0; oS1[grow] = s1;

    unsafeAtomicAdd(&sDen[i0], s0);
    unsafeAtomicAdd(&sDen[i1], s1);
    unsafeAtomicAdd(&sCnt[i0], 1.0f);
    unsafeAtomicAdd(&sCnt[i1], 1.0f);

    // write raw softmax scores back for importance column-sums
    #pragma unroll
    for (int j = 0; j < 16; ++j) {
      v4f q;
      q.x = r[j*4+0] * inv; q.y = r[j*4+1] * inv;
      q.z = r[j*4+2] * inv; q.w = r[j*4+3] * inv;
      *(v4f*)(lrow + j * 4) = q;
    }
  }
  __syncthreads();

  // Importance: 4 partial column sums per expert across 128 rows
  {
    int e = tid & 63, q = tid >> 6;
    float p = 0.f;
    for (int rr = q * 32; rr < q * 32 + 32; ++rr) p += sLog[rr * 68 + e];
    unsafeAtomicAdd(&sImp[e], p);
  }
  __syncthreads();

  if (tid < 64) {
    unsafeAtomicAdd(&gDen[tid], sDen[tid]);
    unsafeAtomicAdd(&gImp[tid], sImp[tid]);
    unsafeAtomicAdd(&gCnt[tid], sCnt[tid]);
  }
}

// ---------------- finalize: per-expert scale factors + aux loss ----------------
__global__ void moe_finalize_kernel(const float* __restrict__ gDen,
                                    const float* __restrict__ gImp,
                                    const float* __restrict__ gCnt,
                                    float* __restrict__ F, float* __restrict__ out_aux) {
  int t = threadIdx.x;
  if (t < 64) F[t] = CAPF / (gDen[t] + 1e-6f);
  if (t == 0) {
    float a = 0.f;
    for (int e = 0; e < 64; ++e) {
      float imp = gImp[e] * (1.0f / (float)NROWS);
      float ld  = gCnt[e] * (1.0f / (float)NROWS);
      float d = ld - imp;
      a += d * d;
    }
    *out_aux = 0.01f * (a * (1.0f / 64.0f));
  }
}

// ---------------- scatter: dense [B,64] output, coalesced ----------------
__global__ __launch_bounds__(256) void moe_scatter_kernel(
    const int* __restrict__ oI0, const int* __restrict__ oI1,
    const float* __restrict__ oS0, const float* __restrict__ oS1,
    const float* __restrict__ F, float* __restrict__ out) {
  int gid = blockIdx.x * 256 + threadIdx.x;
  int row = gid >> 6, e = gid & 63;
  int i0 = oI0[row], i1 = oI1[row];
  float v = (e == i0) ? oS0[row] : (e == i1) ? oS1[row] : 0.0f;
  out[gid] = v * F[e];
}

extern "C" void kernel_launch(void* const* d_in, const int* in_sizes, int n_in,
                              void* d_out, int out_size, void* d_ws, size_t ws_size,
                              hipStream_t stream) {
  (void)in_sizes; (void)n_in; (void)out_size; (void)ws_size;
  const float* x = (const float*)d_in[0];
  const float* W = (const float*)d_in[1];
  const float* b = (const float*)d_in[2];

  char* ws = (char*)d_ws;
  unsigned short* whi = (unsigned short*)ws; ws += (size_t)NEXP * DIMK * 2;
  unsigned short* wlo = (unsigned short*)ws; ws += (size_t)NEXP * DIMK * 2;
  float* accums = (float*)ws;                 // den[64] | imp[64] | cnt[64]
  float* gDen = accums;
  float* gImp = accums + 64;
  float* gCnt = accums + 128;
  ws += 192 * 4;
  float* F   = (float*)ws; ws += 64 * 4;
  int*   oI0 = (int*)ws;   ws += (size_t)NROWS * 4;
  int*   oI1 = (int*)ws;   ws += (size_t)NROWS * 4;
  float* oS0 = (float*)ws; ws += (size_t)NROWS * 4;
  float* oS1 = (float*)ws; ws += (size_t)NROWS * 4;

  float* out = (float*)d_out;

  moe_prep_kernel<<<(NEXP * DIMK) / 256, 256, 0, stream>>>(W, whi, wlo, accums);
  moe_gate_kernel<<<NROWS / 128, 256, 0, stream>>>(x, b, whi, wlo,
                                                   gDen, gImp, gCnt, oI0, oI1, oS0, oS1);
  moe_finalize_kernel<<<1, 64, 0, stream>>>(gDen, gImp, gCnt, F, out + (size_t)NROWS * NEXP);
  moe_scatter_kernel<<<(NROWS * NEXP) / 256, 256, 0, stream>>>(oI0, oI1, oS0, oS1, F, out);
}
